// UniversalMoEBlock_39539468927390
// MI455X (gfx1250) — matrix operand order repaired and checked by reading, not compiled, not run on hardware
//
#include <hip/hip_runtime.h>
#include <hip/hip_bf16.h>

// ---------------------------------------------------------------------------
// Types for CDNA5 WMMA (wave32, 16x16x32 bf16 -> f32 accumulate)
// ---------------------------------------------------------------------------
typedef __attribute__((ext_vector_type(16))) __bf16 v16bf;
typedef __attribute__((ext_vector_type(8)))  __bf16 v8bf;
typedef __attribute__((ext_vector_type(8)))  float  v8f;
typedef __attribute__((ext_vector_type(4)))  float  f4;
typedef __attribute__((ext_vector_type(4)))  unsigned uint32x4;
typedef __attribute__((ext_vector_type(8)))  int      int32x8;
typedef __attribute__((ext_vector_type(4)))  int      int32x4;

union FragBF {
    v16bf v;
    v8bf  h[2];
};

__device__ __forceinline__ v8f wmma_bf16(const FragBF& a, const FragBF& b, v8f c) {
    return __builtin_amdgcn_wmma_f32_16x16x32_bf16(
        /*neg_a=*/false, a.v, /*neg_b=*/false, b.v,
        /*c_mod=*/(short)0, c, /*reuse_a=*/false, /*reuse_b=*/false);
}

__device__ __forceinline__ __bf16 f2bf(float f) {
    union { float f; unsigned u; } a; a.f = f;
    unsigned u = a.u + 0x7FFFu + ((a.u >> 16) & 1u);   // round-to-nearest-even
    union { unsigned short s; __bf16 b; } o; o.s = (unsigned short)(u >> 16);
    return o.b;
}

#if defined(__has_builtin)
#if __has_builtin(__builtin_amdgcn_cvt_pk_bf16_f32)
#define HAVE_CVT_PK_BF16 1
#endif
#if __has_builtin(__builtin_amdgcn_tensor_load_to_lds) && __has_builtin(__builtin_amdgcn_s_wait_tensorcnt)
#define HAVE_TDM 1
#endif
#endif

// store two bf16 (4-byte aligned destination)
__device__ __forceinline__ void store_bf16x2(__bf16* dst, float x, float y) {
#ifdef HAVE_CVT_PK_BF16
    auto p = __builtin_amdgcn_cvt_pk_bf16_f32(x, y);
    __builtin_memcpy(dst, &p, 4);
#else
    dst[0] = f2bf(x); dst[1] = f2bf(y);
#endif
}

// Problem constants
#define BB   2
#define SS   2048
#define HH   1024
#define EE   8
#define RR   128
#define II   4096
#define HR   1024
#define TT   (BB * SS)         // 4096 tokens
#define G3   (3 * HR)          // 3072

// GEMM tiling
#define TM    128
#define TN    64                // gate/up N tile
#define BTN   128               // big kernels N tile
#define TK    32
#define PITCH 48                // bf16 pitch: 96B rows, keeps b128 alignment

// ---------------------------------------------------------------------------
// Async copy to LDS (raw bytes) + waits
// ---------------------------------------------------------------------------
__device__ __forceinline__ void async_b128_to_lds(const void* gp, void* lds) {
    unsigned lds_off = (unsigned)(uintptr_t)lds;   // low 32 bits == LDS offset
    asm volatile("global_load_async_to_lds_b128 %0, %1, off"
                 :: "v"(lds_off), "v"(gp) : "memory");
}
__device__ __forceinline__ void wait_async0() {
    asm volatile("s_wait_asynccnt 0x0" ::: "memory");
}
__device__ __forceinline__ void wait_async1() {
    asm volatile("s_wait_asynccnt 0x1" ::: "memory");
}

#ifdef HAVE_TDM
// Tensor Data Mover: load a rows x cols tile of bf16 from global into LDS.
// LDS pitch = 96B (64B data row + 32B pad): pad_interval=16 DW, pad_amount=8 DW.
__device__ __forceinline__ void tdm_load_bf16_tile(const __bf16* gptr, void* lds_dst,
                                                   int rows, int cols, int stride_elems) {
    unsigned long long ga = (unsigned long long)(uintptr_t)gptr;
    uint32x4 g0 = {0u, 0u, 0u, 0u};
    g0[0] = 1u;                                    // count=1 (valid user descriptor)
    g0[1] = (unsigned)(uintptr_t)lds_dst;          // lds_addr (bytes)
    g0[2] = (unsigned)ga;                          // global_addr[31:0]
    g0[3] = (unsigned)((ga >> 32) & 0x01FFFFFFull) | (2u << 30);  // addr[56:32] | type=2
    int32x8 g1 = {0, 0, 0, 0, 0, 0, 0, 0};
    g1[0] = (int)((1u << 16)     // data_size = 2 bytes
                | (1u << 20)     // pad_enable
                | (3u << 22)     // pad_interval: 16 DWORDs (= one 32-elem bf16 row)
                | (7u << 25));   // pad_amount:   8 DWORDs (= 32B -> 96B pitch)
    unsigned td0 = (unsigned)stride_elems;         // tensor_dim0 (elements)
    g1[1] = (int)((td0 & 0xFFFFu) << 16);                          // td0 lo16
    g1[2] = (int)(((td0 >> 16) & 0xFFFFu) | (0xFFFFu << 16));      // td0 hi16 | td1 lo16
    g1[3] = (int)(((unsigned)cols & 0xFFFFu) << 16);               // td1 hi16=0 | tile_dim0
    g1[4] = (int)((unsigned)rows & 0xFFFFu);                       // tile_dim1 | tile_dim2=0
    g1[5] = (int)(unsigned)stride_elems;                           // tensor_dim0_stride lo32
    int32x4 z4 = {0, 0, 0, 0};
#if defined(__clang_major__) && __clang_major__ >= 23
    int32x8 z8 = {0, 0, 0, 0, 0, 0, 0, 0};
    __builtin_amdgcn_tensor_load_to_lds(g0, g1, z4, z4, z8, 0);
#else
    __builtin_amdgcn_tensor_load_to_lds(g0, g1, z4, z4, 0);
#endif
}
#endif

// ---------------------------------------------------------------------------
// Register-staged tile movers (two-phase for software pipelining)
// ---------------------------------------------------------------------------
struct RegsA { f4 v[4]; };                 // 128 x 32 fp32 tile
template <int N> struct RegsF { f4 v[N]; };

__device__ __forceinline__ void loadA_f32(RegsA& r, const float* __restrict__ A, int lda,
                                          int m0, int k0, int tid) {
#pragma unroll
    for (int i = 0; i < 4; ++i) {
        int idx = tid + i * 256;
        int row = idx >> 3, c = (idx & 7) << 2;
        r.v[i] = *(const f4*)(A + (size_t)(m0 + row) * lda + k0 + c);
    }
}
__device__ __forceinline__ void storeA_bf16(const RegsA& r, __bf16 (*Abuf)[PITCH], int tid) {
#pragma unroll
    for (int i = 0; i < 4; ++i) {
        int idx = tid + i * 256;
        int row = idx >> 3, c = (idx & 7) << 2;
        store_bf16x2(&Abuf[row][c],     r.v[i].x, r.v[i].y);
        store_bf16x2(&Abuf[row][c + 2], r.v[i].z, r.v[i].w);
    }
}

// "BT": B row-major [N,K]; tile NROWS x 32
template <int ITERS>
__device__ __forceinline__ void loadBT(RegsF<ITERS>& r, const float* __restrict__ B, int ldb,
                                       int n0, int k0, int tid) {
#pragma unroll
    for (int i = 0; i < ITERS; ++i) {
        int idx = tid + i * 256;
        int row = idx >> 3, c = (idx & 7) << 2;
        r.v[i] = __builtin_nontemporal_load((const f4*)(B + (size_t)(n0 + row) * ldb + k0 + c));
    }
}
template <int ITERS>
__device__ __forceinline__ void storeBT(const RegsF<ITERS>& r, __bf16 (*Bbuf)[PITCH], int tid) {
#pragma unroll
    for (int i = 0; i < ITERS; ++i) {
        int idx = tid + i * 256;
        int row = idx >> 3, c = (idx & 7) << 2;
        store_bf16x2(&Bbuf[row][c],     r.v[i].x, r.v[i].y);
        store_bf16x2(&Bbuf[row][c + 2], r.v[i].z, r.v[i].w);
    }
}

// "BN": B row-major [K,N]; tile 32 x NCOLS, stored transposed into LDS [n][k]
template <int ITERS, int CSHIFT>
__device__ __forceinline__ void loadBN(RegsF<ITERS>& r, const float* __restrict__ B, int ldb,
                                       int n0, int k0, int tid) {
#pragma unroll
    for (int i = 0; i < ITERS; ++i) {
        int idx = tid + i * 256;
        int k = idx >> CSHIFT, c = (idx & ((1 << CSHIFT) - 1)) << 2;
        r.v[i] = __builtin_nontemporal_load((const f4*)(B + (size_t)(k0 + k) * ldb + n0 + c));
    }
}
template <int ITERS, int CSHIFT>
__device__ __forceinline__ void storeBN(const RegsF<ITERS>& r, __bf16 (*Bbuf)[PITCH], int tid) {
#pragma unroll
    for (int i = 0; i < ITERS; ++i) {
        int idx = tid + i * 256;
        int k = idx >> CSHIFT, c = (idx & ((1 << CSHIFT) - 1)) << 2;
        Bbuf[c + 0][k] = f2bf(r.v[i].x);
        Bbuf[c + 1][k] = f2bf(r.v[i].y);
        Bbuf[c + 2][k] = f2bf(r.v[i].z);
        Bbuf[c + 3][k] = f2bf(r.v[i].w);
    }
}

// Fragment loads from LDS
__device__ __forceinline__ void load_a_frag(FragBF& a, __bf16 (*Abuf)[PITCH],
                                            int row_base, int l15, int half) {
    a.h[0] = *(const v8bf*)&Abuf[row_base + l15][half * 8];
    a.h[1] = *(const v8bf*)&Abuf[row_base + l15][16 + half * 8];
}
__device__ __forceinline__ void load_b_frag(FragBF& b, __bf16 (*Bbuf)[PITCH],
                                            int col_base, int l15, int half) {
    b.h[0] = *(const v8bf*)&Bbuf[col_base + l15][half * 8];
    b.h[1] = *(const v8bf*)&Bbuf[col_base + l15][16 + half * 8];
}

// ---------------------------------------------------------------------------
// Kernel 1: C[M,N] = A[M,K] * B^T   (B stored [N,K]),  fp32 in / fp32 out
// WG tile 128x128, wave tile 32x64, software-pipelined (regs double buffer).
// grid = (N/128, M/128), block = 256
// ---------------------------------------------------------------------------
__global__ void __launch_bounds__(256)
gemm_bt_kernel(const float* __restrict__ A, const float* __restrict__ B,
               float* __restrict__ C, int Kdim, int lda, int ldb, int ldc) {
    __shared__ __bf16 Abuf[TM][PITCH];
    __shared__ __bf16 Bbuf[BTN][PITCH];

    const int tid  = threadIdx.x;
    const int lane = tid & 31, wv = tid >> 5;
    const int half = lane >> 4, l15 = lane & 15;
    const int wm = wv >> 1, wn = wv & 1;
    const int m0 = blockIdx.y * TM, n0 = blockIdx.x * BTN;

    v8f acc[2][4] = {};
    RegsA ra;
    RegsF<4> rb;
    loadA_f32(ra, A, lda, m0, 0, tid);
    loadBT<4>(rb, B, ldb, n0, 0, tid);

    for (int kt = 0; kt < Kdim; kt += TK) {
        storeA_bf16(ra, Abuf, tid);
        storeBT<4>(rb, Bbuf, tid);
        __syncthreads();
        const bool nxt = (kt + TK) < Kdim;
        if (nxt) {                          // prefetch next tile into registers
            loadA_f32(ra, A, lda, m0, kt + TK, tid);
            loadBT<4>(rb, B, ldb, n0, kt + TK, tid);
        }
        if (kt + 2 * TK < Kdim)
            __builtin_prefetch(A + (size_t)(m0 + (tid >> 3)) * lda + kt + 2 * TK, 0, 1);

        FragBF a[2], b[4];
        load_a_frag(a[0], Abuf, wm * 32,      l15, half);
        load_a_frag(a[1], Abuf, wm * 32 + 16, l15, half);
#pragma unroll
        for (int j = 0; j < 4; ++j)
            load_b_frag(b[j], Bbuf, wn * 64 + j * 16, l15, half);
#pragma unroll
        for (int i = 0; i < 2; ++i)
#pragma unroll
            for (int j = 0; j < 4; ++j)
                acc[i][j] = wmma_bf16(a[i], b[j], acc[i][j]);
        __syncthreads();
    }

#pragma unroll
    for (int i = 0; i < 2; ++i)
#pragma unroll
        for (int j = 0; j < 4; ++j)
#pragma unroll
            for (int r = 0; r < 8; ++r) {
                int m = m0 + wm * 32 + i * 16 + half * 8 + r;
                int n = n0 + wn * 64 + j * 16 + l15;
                C[(size_t)m * ldc + n] = acc[i][j][r];
            }
}

// ---------------------------------------------------------------------------
// Kernel 2: fused gate/up:  T1 = bf16( silu(X@Wg) * (X@Wu) ), pipelined.
// WG tile 128x64, wave tile 32x32 (8 WMMAs/step). grid = (N/64, M/128)
// ---------------------------------------------------------------------------
__global__ void __launch_bounds__(256)
moe_gateup_kernel(const float* __restrict__ X, const float* __restrict__ Wg,
                  const float* __restrict__ Wu, __bf16* __restrict__ T1,
                  int Kdim, int lda, int ldb, int ldt) {
    __shared__ __bf16 Abuf[TM][PITCH];
    __shared__ __bf16 Gbuf[TN][PITCH];
    __shared__ __bf16 Ubuf[TN][PITCH];

    const int tid  = threadIdx.x;
    const int lane = tid & 31, wv = tid >> 5;
    const int half = lane >> 4, l15 = lane & 15;
    const int wm = wv >> 1, wn = wv & 1;
    const int m0 = blockIdx.y * TM, n0 = blockIdx.x * TN;

    v8f g[2][2] = {}, u[2][2] = {};
    RegsA ra;
    RegsF<2> rg, ru;
    loadA_f32(ra, X, lda, m0, 0, tid);
    loadBN<2, 4>(rg, Wg, ldb, n0, 0, tid);
    loadBN<2, 4>(ru, Wu, ldb, n0, 0, tid);

    for (int kt = 0; kt < Kdim; kt += TK) {
        storeA_bf16(ra, Abuf, tid);
        storeBN<2, 4>(rg, Gbuf, tid);
        storeBN<2, 4>(ru, Ubuf, tid);
        __syncthreads();
        const bool nxt = (kt + TK) < Kdim;
        if (nxt) {
            loadA_f32(ra, X, lda, m0, kt + TK, tid);
            loadBN<2, 4>(rg, Wg, ldb, n0, kt + TK, tid);
            loadBN<2, 4>(ru, Wu, ldb, n0, kt + TK, tid);
        }

        FragBF a[2], bg[2], bu[2];
        load_a_frag(a[0], Abuf, wm * 32,      l15, half);
        load_a_frag(a[1], Abuf, wm * 32 + 16, l15, half);
#pragma unroll
        for (int j = 0; j < 2; ++j) {
            load_b_frag(bg[j], Gbuf, wn * 32 + j * 16, l15, half);
            load_b_frag(bu[j], Ubuf, wn * 32 + j * 16, l15, half);
        }
#pragma unroll
        for (int i = 0; i < 2; ++i)
#pragma unroll
            for (int j = 0; j < 2; ++j) {
                g[i][j] = wmma_bf16(a[i], bg[j], g[i][j]);
                u[i][j] = wmma_bf16(a[i], bu[j], u[i][j]);
            }
        __syncthreads();
    }

#pragma unroll
    for (int i = 0; i < 2; ++i)
#pragma unroll
        for (int j = 0; j < 2; ++j)
#pragma unroll
            for (int r = 0; r < 8; ++r) {
                int m = m0 + wm * 32 + i * 16 + half * 8 + r;
                int n = n0 + wn * 32 + j * 16 + l15;
                float gv = g[i][j][r], uv = u[i][j][r];
                float s = gv / (1.f + __expf(-gv)) * uv;
                T1[(size_t)m * ldt + n] = f2bf(s);
            }
}

// ---------------------------------------------------------------------------
// Kernel 3: down projection: Out[m,n] (+)= w[m,e] * (T1 @ Wd)[m,n]
// A tiles (bf16) via TDM double-buffer (s_wait_tensorcnt pipelining);
// Wd fp32 "BN"; WG tile 128x128, wave tile 32x64. grid = (N/128, M/128)
// ---------------------------------------------------------------------------
__global__ void __launch_bounds__(256)
moe_down_kernel(const __bf16* __restrict__ T1, const float* __restrict__ Wd,
                const float* __restrict__ Wt, int eidx, float* __restrict__ Out,
                int Kdim, int lda, int ldb, int ldc, int accumulate) {
    __shared__ __bf16 Ab[2][TM][PITCH];
    __shared__ __bf16 Bbuf[BTN][PITCH];

    const int tid  = threadIdx.x;
    const int lane = tid & 31, wv = tid >> 5;
    const int half = lane >> 4, l15 = lane & 15;
    const int wm = wv >> 1, wn = wv & 1;
    const int m0 = blockIdx.y * TM, n0 = blockIdx.x * BTN;

    v8f acc[2][4] = {};
    RegsF<4> rb;
    loadBN<4, 5>(rb, Wd, ldb, n0, 0, tid);

#ifdef HAVE_TDM
    if (tid < 32)
        tdm_load_bf16_tile(T1 + (size_t)m0 * lda, &Ab[0][0][0], TM, TK, lda);
#endif

    int cur = 0;
    for (int kt = 0; kt < Kdim; kt += TK) {
        const bool nxt = (kt + TK) < Kdim;
        storeBN<4, 5>(rb, Bbuf, tid);
#ifdef HAVE_TDM
        if (tid < 32) {
            if (nxt) {   // issue DMA for next tile, then retire current (<=1 in flight)
                tdm_load_bf16_tile(T1 + (size_t)m0 * lda + kt + TK, &Ab[cur ^ 1][0][0], TM, TK, lda);
                __builtin_amdgcn_s_wait_tensorcnt(1);
            } else {
                __builtin_amdgcn_s_wait_tensorcnt(0);
            }
        }
#else
        {   // async fallback: raw b128 copies into current buffer
#pragma unroll
            for (int i = 0; i < 2; ++i) {
                int idx = tid + i * 256;
                int row = idx >> 2, c8 = (idx & 3) << 3;
                async_b128_to_lds(T1 + (size_t)(m0 + row) * lda + kt + c8, &Ab[cur][row][c8]);
            }
            wait_async0();
        }
#endif
        __syncthreads();
        if (nxt) loadBN<4, 5>(rb, Wd, ldb, n0, kt + TK, tid);

        FragBF a[2], b[4];
        load_a_frag(a[0], Ab[cur], wm * 32,      l15, half);
        load_a_frag(a[1], Ab[cur], wm * 32 + 16, l15, half);
#pragma unroll
        for (int j = 0; j < 4; ++j)
            load_b_frag(b[j], Bbuf, wn * 64 + j * 16, l15, half);
#pragma unroll
        for (int i = 0; i < 2; ++i)
#pragma unroll
            for (int j = 0; j < 4; ++j)
                acc[i][j] = wmma_bf16(a[i], b[j], acc[i][j]);
        __syncthreads();
        cur ^= 1;
    }

#pragma unroll
    for (int i = 0; i < 2; ++i)
#pragma unroll
        for (int j = 0; j < 4; ++j)
#pragma unroll
            for (int r = 0; r < 8; ++r) {
                int m = m0 + wm * 32 + i * 16 + half * 8 + r;
                int n = n0 + wn * 64 + j * 16 + l15;
                float scale = Wt ? Wt[m * EE + eidx] : 1.0f;
                float v = scale * acc[i][j][r];
                size_t o = (size_t)m * ldc + n;
                if (accumulate) Out[o] += v; else Out[o] = v;
            }
}

// ---------------------------------------------------------------------------
// Kernel 4: persistent GRU scan. 32 workgroups, each owns a 32-column slice.
// Next step's xw gate rows prefetched into LDS via async-to-LDS DMA.
// ---------------------------------------------------------------------------
#define NWG_GRU 32
__global__ void __launch_bounds__(256)
gru_scan_kernel(const float* __restrict__ XW, const float* __restrict__ Whh,
                float* __restrict__ hbuf, float* __restrict__ RSEQ,
                unsigned* __restrict__ bar) {
    __shared__ float hl[BB * HR];
    __shared__ float dots[BB * 3 * 32];
    __shared__ float xwb[2][BB * 3 * 32];   // async-prefetched xw rows

    const int tid  = threadIdx.x;
    const int lane = tid & 31, wv = tid >> 5;
    const int wg   = blockIdx.x;            // 0..31

    // prefetch xw strip for step 0
    if (tid < 48) {
        int b = tid / 24, rem = tid % 24;
        int gate = rem >> 3, q = rem & 7;
        async_b128_to_lds(XW + (size_t)(b * SS) * G3 + gate * HR + wg * 32 + q * 4,
                          &xwb[0][b * 96 + gate * 32 + q * 4]);
    }

    for (int s = 0; s < SS; ++s) {
        const int cur = s & 1, nxt = cur ^ 1;
        const bool more = (s + 1) < SS;
        for (int i = tid; i < BB * HR; i += 256) hl[i] = hbuf[cur * BB * HR + i];
        // prefetch next step's xw strip while we do the matvec
        if (more && tid < 48) {
            int b = tid / 24, rem = tid % 24;
            int gate = rem >> 3, q = rem & 7;
            async_b128_to_lds(XW + (size_t)(b * SS + s + 1) * G3 + gate * HR + wg * 32 + q * 4,
                              &xwb[nxt][b * 96 + gate * 32 + q * 4]);
        }
        __syncthreads();

        // 192 dot products of length 1024, one wave each
        for (int d = wv; d < BB * 3 * 32; d += 8) {
            int b = d / 96; int rem = d - b * 96;
            int gate = rem >> 5; int jj = rem & 31;
            int row = gate * HR + wg * 32 + jj;
            const float* wr = Whh + (size_t)row * HR;
            const float* hb = hl + b * HR;
            float acc = 0.f;
#pragma unroll 8
            for (int h = lane; h < HR; h += 32) acc += hb[h] * wr[h];
#pragma unroll
            for (int m = 16; m; m >>= 1) acc += __shfl_xor(acc, m, 32);
            if (lane == 0) dots[d] = acc;
        }
        __syncthreads();

        // elementwise GRU update (waves 0-1 issued the asyncs; wait leaves the
        // next-step prefetch in flight)
        if (tid < BB * 32) {
            if (more) wait_async1(); else wait_async0();
            int b = tid >> 5, jj = tid & 31;
            int j = wg * 32 + jj;
            int t = b * SS + s;
            float xr = xwb[cur][b * 96 + jj];
            float xz = xwb[cur][b * 96 + 32 + jj];
            float xn = xwb[cur][b * 96 + 64 + jj];
            float hr = dots[b * 96 + jj];
            float hz = dots[b * 96 + 32 + jj];
            float hn = dots[b * 96 + 64 + jj];
            float r = 1.f / (1.f + __expf(-(xr + hr)));
            float z = 1.f / (1.f + __expf(-(xz + hz)));
            float n = tanhf(xn + r * hn);
            float ho = hl[b * HR + j];
            float hnew = (1.f - z) * n + z * ho;
            hbuf[nxt * BB * HR + b * HR + j] = hnew;
            RSEQ[(size_t)t * HR + j] = hnew;
        }
        __syncthreads();

        // device-wide barrier: monotonic counter, no reset needed
        if (tid == 0) {
            __threadfence();
            atomicAdd(bar, 1u);
            unsigned target = (unsigned)(s + 1) * NWG_GRU;
            while (atomicAdd(bar, 0u) < target)
                __builtin_amdgcn_s_sleep(2);
            __threadfence();
        }
        __syncthreads();
    }
}

// ---------------------------------------------------------------------------
// Kernel 5: routing stats per token: l2norm, gram, penalty partial, cosines
// grid = 4096 (one block per token), block = 256 (8 waves = 8 experts)
// ---------------------------------------------------------------------------
__global__ void __launch_bounds__(256)
routing_stats_kernel(const float* __restrict__ RSEQ, const float* __restrict__ EXPR,
                     float* __restrict__ COS, float* __restrict__ pen_sum) {
    __shared__ float rn[HR];
    __shared__ float gram[EE * EE];

    const int t = blockIdx.x;
    const int tid = threadIdx.x, lane = tid & 31, wv = tid >> 5;   // wv == expert
    const float* rt = RSEQ + (size_t)t * HR;

    {
        float a = 0.f;
        for (int i = lane; i < RR; i += 32) { float v = rt[wv * RR + i]; a += v * v; }
#pragma unroll
        for (int m = 16; m; m >>= 1) a += __shfl_xor(a, m, 32);
        float inv = 1.f / fmaxf(sqrtf(a), 1e-12f);
        for (int i = lane; i < RR; i += 32) rn[wv * RR + i] = rt[wv * RR + i] * inv;
    }
    __syncthreads();

    for (int f = 0; f < EE; ++f) {
        float a = 0.f;
        for (int i = lane; i < RR; i += 32) a += rn[wv * RR + i] * rn[f * RR + i];
#pragma unroll
        for (int m = 16; m; m >>= 1) a += __shfl_xor(a, m, 32);
        if (lane == 0) gram[wv * EE + f] = a;
    }
    __syncthreads();

    if (tid == 0) {
        float pt = 0.f;
        for (int e = 0; e < EE; ++e) {
            float s2 = 0.f;
            for (int f = 0; f < EE; ++f) {
                float d = gram[e * EE + f] - (e == f ? 1.f : 0.f);
                s2 += d * d;
            }
            float nrm = fmaxf(sqrtf(s2), 1e-12f);
            pt += s2 / (nrm * nrm);
        }
        atomicAdd(pen_sum, pt);
    }

    {
        const float* ex = EXPR + (size_t)t * HR + wv * RR;
        float dot = 0.f, na = 0.f, nb = 0.f;
        for (int i = lane; i < RR; i += 32) {
            float a_ = ex[i], b_ = rn[wv * RR + i];
            dot += a_ * b_; na += a_ * a_; nb += b_ * b_;
        }
#pragma unroll
        for (int m = 16; m; m >>= 1) {
            dot += __shfl_xor(dot, m, 32);
            na  += __shfl_xor(na,  m, 32);
            nb  += __shfl_xor(nb,  m, 32);
        }
        if (lane == 0)
            COS[t * EE + wv] = 1.f - dot / (fmaxf(sqrtf(na), 1e-8f) * fmaxf(sqrtf(nb), 1e-8f));
    }
}

// ---------------------------------------------------------------------------
// Kernel 6: top-2 + softmax -> dense per-token expert weights [T, E]
// ---------------------------------------------------------------------------
__global__ void __launch_bounds__(256)
weights_kernel(const float* __restrict__ COS, const float* __restrict__ pen_sum,
               float* __restrict__ W) {
    int t = blockIdx.x * blockDim.x + threadIdx.x;
    if (t >= TT) return;
    float pen = (*pen_sum) / (float)TT;
    float d[EE];
#pragma unroll
    for (int e = 0; e < EE; ++e) d[e] = COS[t * EE + e] * (1.f + pen);

    int i1 = 0; float v1 = d[0];
#pragma unroll
    for (int e = 1; e < EE; ++e) if (d[e] > v1) { v1 = d[e]; i1 = e; }
    int i2 = -1; float v2 = -3.4e38f;
#pragma unroll
    for (int e = 0; e < EE; ++e) if (e != i1 && d[e] > v2) { v2 = d[e]; i2 = e; }

    float mx = fmaxf(v1, v2);
    float e1 = __expf(v1 - mx), e2 = __expf(v2 - mx);
    float inv = 1.f / (e1 + e2);
#pragma unroll
    for (int e = 0; e < EE; ++e) W[t * EE + e] = 0.f;
    W[t * EE + i1] = e1 * inv;
    W[t * EE + i2] = e2 * inv;
}

// ---------------------------------------------------------------------------
// Launcher
// ---------------------------------------------------------------------------
extern "C" void kernel_launch(void* const* d_in, const int* in_sizes, int n_in,
                              void* d_out, int out_size, void* d_ws, size_t ws_size,
                              hipStream_t stream) {
    (void)in_sizes; (void)n_in; (void)out_size; (void)ws_size;

    const float* X    = (const float*)d_in[0];
    const float* H0   = (const float*)d_in[1];
    const float* Wih  = (const float*)d_in[2];
    const float* Whh  = (const float*)d_in[3];
    const float* Wexp = (const float*)d_in[4];
    const float* Wg   = (const float*)d_in[5];
    const float* Wu   = (const float*)d_in[6];
    const float* Wd   = (const float*)d_in[7];
    const float* SWg  = (const float*)d_in[8];
    const float* SWu  = (const float*)d_in[9];
    const float* SWd  = (const float*)d_in[10];
    float* OUT = (float*)d_out;

    char* ws = (char*)d_ws;
    float*  XW   = (float*)ws;  ws += (size_t)TT * G3 * 4;      // 48 MB
    float*  EXPR = (float*)ws;  ws += (size_t)TT * HR * 4;      // 16 MB
    float*  RSEQ = (float*)ws;  ws += (size_t)TT * HR * 4;      // 16 MB
    __bf16* T1   = (__bf16*)ws; ws += (size_t)TT * II * 2;      // 32 MB
    float*  HBUF = (float*)ws;  ws += 2 * BB * HR * 4;
    float*  COS  = (float*)ws;  ws += TT * EE * 4;
    float*  WTS  = (float*)ws;  ws += TT * EE * 4;
    float*  PEN  = (float*)ws;  ws += 256;
    unsigned* BAR = (unsigned*)ws; ws += 256;

    hipMemsetAsync(PEN, 0, sizeof(float), stream);
    hipMemsetAsync(BAR, 0, sizeof(unsigned), stream);
    hipMemcpyAsync(HBUF, H0, (size_t)BB * HR * sizeof(float),
                   hipMemcpyDeviceToDevice, stream);

    const dim3 blk(256);

    gemm_bt_kernel<<<dim3(G3 / BTN, TT / TM), blk, 0, stream>>>(
        X, Wih, XW, HH, HH, HH, G3);
    gemm_bt_kernel<<<dim3(HR / BTN, TT / TM), blk, 0, stream>>>(
        X, Wexp, EXPR, HH, HH, HH, HR);

    gru_scan_kernel<<<dim3(NWG_GRU), blk, 0, stream>>>(XW, Whh, HBUF, RSEQ, BAR);

    routing_stats_kernel<<<dim3(TT), blk, 0, stream>>>(RSEQ, EXPR, COS, PEN);
    weights_kernel<<<dim3(TT / 256), blk, 0, stream>>>(COS, PEN, WTS);

    moe_gateup_kernel<<<dim3(II / TN, TT / TM), blk, 0, stream>>>(
        X, SWg, SWu, T1, HH, HH, II, II);
    moe_down_kernel<<<dim3(HH / BTN, TT / TM), blk, 0, stream>>>(
        T1, SWd, nullptr, 0, OUT, II, II, HH, HH, 0);

    for (int e = 0; e < EE; ++e) {
        const float* Wg_e = Wg + (size_t)e * HH * II;
        const float* Wu_e = Wu + (size_t)e * HH * II;
        const float* Wd_e = Wd + (size_t)e * II * HH;
        moe_gateup_kernel<<<dim3(II / TN, TT / TM), blk, 0, stream>>>(
            X, Wg_e, Wu_e, T1, HH, HH, II, II);
        moe_down_kernel<<<dim3(HH / BTN, TT / TM), blk, 0, stream>>>(
            T1, Wd_e, WTS, e, OUT, II, II, HH, HH, 1);
    }
}
